// ResMPS_basic_68650757259700
// MI455X (gfx1250) — matrix operand order repaired and checked by reading, not compiled, not run on hardware
//
#include <hip/hip_runtime.h>
#include <hip/hip_bf16.h>

// ResMPS on MI455X (gfx1250, wave32, WMMA).
//
// Shapes: x(1024,2,784) f32, W(784,2,128,128) f32, V(128,10) f32, out(1024,10) f32.
//   Kernel 1: W -> bf16, pre-swizzled into WMMA B-fragment order in d_ws (51.4 MB,
//             L2-resident on a 192MB-L2 part; HBM reads W once).
//   Kernel 2: 64 blocks x 256 threads (8 waves). Block owns batch rows [16b,16b+16);
//             wave w keeps H columns [16w,16w+16) as f32 v8f accumulators across all
//             784 sites. Per site: 8x v_wmma_f32_16x16x32_bf16 per wave in TWO
//             independent split-K chains per spin (depth 2, hides WMMA hazards),
//             bf16 H exchanged via double-buffered LDS A-fragments (1 barrier/site).

typedef __attribute__((ext_vector_type(16))) __bf16 v16bf;
typedef __attribute__((ext_vector_type(8)))  float  v8f;

#define LENGTH   784
#define DSPIN    2
#define CHI      128
#define CLASSES  10
#define BATCH    1024
#define BT       16                  // batch rows per workgroup
#define NKC      4                   // K chunks of 32 (CHI=128)
#define NJT      8                   // 16-wide column tiles (CHI=128)
#define FRAG_EL  512                 // 32 lanes * 16 bf16 per B fragment
#define SITE_EL  (DSPIN * NKC * NJT * FRAG_EL)   // 32768 bf16 elems = 64KB per site
#define PF_DIST  4                   // prefetch lookahead in sites

__device__ __forceinline__ unsigned short f2bf(float f) {
    union { float f; unsigned u; } v; v.f = f;
    unsigned u = v.u + 0x7FFFu + ((v.u >> 16) & 1u);   // round-to-nearest-even
    return (unsigned short)(u >> 16);
}

// ---------------------------------------------------------------------------
// Kernel 1: W (L,d,chi,chi) f32  ->  bf16 B-fragments [n][s][kc][jt][lane][e]
// B-fragment layout packed here (contiguous K; A uses same order): lane<16 : N=lane,    K'=e      (e=0..15)
//                                     lane>=16: N=lane-16, K'=16+e
// ---------------------------------------------------------------------------
__global__ __launch_bounds__(256) void swizzle_w_kernel(
    const float* __restrict__ W, unsigned short* __restrict__ Wbf)
{
    size_t t = (size_t)blockIdx.x * 256 + threadIdx.x;
    const size_t total = (size_t)LENGTH * SITE_EL;
    if (t >= total) return;
    int e    = (int)(t        & 15);
    int lane = (int)((t >> 4) & 31);
    int jt   = (int)((t >> 9) & 7);
    int kc   = (int)((t >> 12) & 3);
    int s    = (int)((t >> 14) & 1);
    int n    = (int)(t >> 15);
    int a = kc * 32 + ((lane & 16) ? (16 + e) : e);     // K index (row of W[n,s])
    int j = jt * 16 + (lane & 15);                      // N index (col of W[n,s])
    size_t src = (((size_t)n * DSPIN + s) * CHI + a) * CHI + j;
    Wbf[t] = f2bf(W[src]);
}

// ---------------------------------------------------------------------------
// Kernel 2: the sequential site recurrence + classifier
// ---------------------------------------------------------------------------
__global__ __launch_bounds__(256) void resmps_kernel(
    const float* __restrict__ x,
    const unsigned short* __restrict__ Wbf,
    const float* __restrict__ V,
    float* __restrict__ out)
{
    const int tid  = threadIdx.x;
    const int w    = tid >> 5;          // wave id 0..7 -> H column tile
    const int lane = tid & 31;
    const int b0   = blockIdx.x * BT;   // first batch row of this block

    // Double-buffered (by site parity) A-fragments + x scalars: 1 barrier/site.
    __shared__ alignas(32) unsigned short Afrag[2][NKC * 32 * 16]; // 2 x 4KB
    __shared__ float xs[2][DSPIN][BT];
    __shared__ float Hfin[BT][CHI];                                // epilogue tile

    // C/D fragment row mapping: VGPR r holds M = r + (lane>=16 ? 8 : 0)
    const int mbase = (lane & 16) ? 8 : 0;
    // H column owned by this lane (lanes l and l+16 share a column, rows differ)
    const int jcol = (w << 4) + (lane & 15);
    // Scatter target inside the A-fragment buffer for column jcol (K' = jcol%32):
    const int kc_j = jcol >> 5;
    const int Kp   = jcol & 31;
    const int lh   = (Kp >> 4) & 1;                       // high/low lane group
    const int ee   = Kp & 15;                             // element within v16bf
    const int scat = (((kc_j << 5) + (lh << 4)) << 4) + ee;  // + (m<<4)

    // H accumulator: h0 = ones
    float accH[8];
#pragma unroll
    for (int r = 0; r < 8; ++r) accH[r] = 1.0f;

    // ---- prologue: publish site-0 operands ---------------------------------
#pragma unroll
    for (int r = 0; r < 8; ++r)
        Afrag[0][scat + ((r + mbase) << 4)] = f2bf(accH[r]);
    if (tid < DSPIN * BT) {
        int sidx = tid >> 4, m = tid & 15;
        xs[0][sidx][m] = x[((size_t)(b0 + m) * DSPIN + sidx) * LENGTH + 0];
    }
    __syncthreads();

    for (int n = 0; n < LENGTH; ++n) {
        const int p = n & 1;

        // ---- y_s = H @ W[n,s] via 16x16x32 bf16 WMMA, split-K (depth-2 chains)
        const __bf16* Wn = (const __bf16*)Wbf + (size_t)n * SITE_EL;

        v16bf a0 = *(const v16bf*)&Afrag[p][((0 << 5) + lane) << 4];
        v16bf a1 = *(const v16bf*)&Afrag[p][((1 << 5) + lane) << 4];
        v16bf a2 = *(const v16bf*)&Afrag[p][((2 << 5) + lane) << 4];
        v16bf a3 = *(const v16bf*)&Afrag[p][((3 << 5) + lane) << 4];

        const __bf16* Wb = Wn + ((size_t)w << 9) + (lane << 4);
#define BFRAG(s_, kc_) (*(const v16bf*)(Wb + (size_t)(((s_) * NKC + (kc_)) * NJT) * FRAG_EL))
        v8f y0a = __builtin_amdgcn_wmma_f32_16x16x32_bf16(
                      false, a0, false, BFRAG(0, 0), (short)0, (v8f){}, false, false);
        v8f y1a = __builtin_amdgcn_wmma_f32_16x16x32_bf16(
                      false, a0, false, BFRAG(1, 0), (short)0, (v8f){}, false, false);
        v8f y0b = __builtin_amdgcn_wmma_f32_16x16x32_bf16(
                      false, a1, false, BFRAG(0, 1), (short)0, (v8f){}, false, false);
        v8f y1b = __builtin_amdgcn_wmma_f32_16x16x32_bf16(
                      false, a1, false, BFRAG(1, 1), (short)0, (v8f){}, false, false);
        y0a = __builtin_amdgcn_wmma_f32_16x16x32_bf16(
                  false, a2, false, BFRAG(0, 2), (short)0, y0a, false, false);
        y1a = __builtin_amdgcn_wmma_f32_16x16x32_bf16(
                  false, a2, false, BFRAG(1, 2), (short)0, y1a, false, false);
        y0b = __builtin_amdgcn_wmma_f32_16x16x32_bf16(
                  false, a3, false, BFRAG(0, 3), (short)0, y0b, false, false);
        y1b = __builtin_amdgcn_wmma_f32_16x16x32_bf16(
                  false, a3, false, BFRAG(1, 3), (short)0, y1b, false, false);
#undef BFRAG

        // ---- residual update: h += x0*(y0a+y0b) + x1*(y1a+y1b) -------------
#pragma unroll
        for (int r = 0; r < 8; ++r) {
            int m = r + mbase;
            accH[r] += xs[p][0][m] * (y0a[r] + y0b[r])
                     + xs[p][1][m] * (y1a[r] + y1b[r]);
        }

        // ---- publish site n+1 operands into the other buffer ---------------
        if (n + 1 < LENGTH) {
#pragma unroll
            for (int r = 0; r < 8; ++r)
                Afrag[p ^ 1][scat + ((r + mbase) << 4)] = f2bf(accH[r]);
            if (tid < DSPIN * BT) {
                int sidx = tid >> 4, m = tid & 15;
                xs[p ^ 1][sidx][m] =
                    x[((size_t)(b0 + m) * DSPIN + sidx) * LENGTH + (n + 1)];
            }
        }

        // L2 prefetch of a future site's 64KB fragment block; W is shared by all
        // blocks (L2-resident), so only block 0 issues the prefetch stream.
        if (blockIdx.x == 0 && n + PF_DIST < LENGTH)
            __builtin_prefetch((const void*)(Wn + (size_t)PF_DIST * SITE_EL +
                                             (size_t)tid * 128), 0, 0);

        __syncthreads();   // single barrier per site (double-buffered exchange)
    }

    // ---- epilogue: logits = H @ V ------------------------------------------
#pragma unroll
    for (int r = 0; r < 8; ++r)
        Hfin[r + mbase][jcol] = accH[r];
    __syncthreads();

    if (tid < BT * CLASSES) {
        int m = tid / CLASSES, c = tid % CLASSES;
        float sum = 0.0f;
#pragma unroll 4
        for (int j = 0; j < CHI; ++j)
            sum += Hfin[m][j] * V[j * CLASSES + c];
        out[(size_t)(b0 + m) * CLASSES + c] = sum;
    }
}

// ---------------------------------------------------------------------------
extern "C" void kernel_launch(void* const* d_in, const int* in_sizes, int n_in,
                              void* d_out, int out_size, void* d_ws, size_t ws_size,
                              hipStream_t stream) {
    const float* x = (const float*)d_in[0];   // (1024, 2, 784)
    const float* W = (const float*)d_in[1];   // (784, 2, 128, 128)
    const float* V = (const float*)d_in[2];   // (128, 10)
    float* out     = (float*)d_out;           // (1024, 10)
    unsigned short* Wbf = (unsigned short*)d_ws;  // 51.4 MB bf16 swizzled weights

    const size_t total = (size_t)LENGTH * SITE_EL;            // 25.7M elements
    const int blocks1 = (int)((total + 255) / 256);
    swizzle_w_kernel<<<blocks1, 256, 0, stream>>>(W, Wbf);    // stream-ordered

    resmps_kernel<<<BATCH / BT, 256, 0, stream>>>(x, Wbf, V, out);
    (void)in_sizes; (void)n_in; (void)out_size; (void)ws_size;
}
